// MemoryGate_34222299415081
// MI455X (gfx1250) — compile-verified, matrix operand order checked by reading, not run on hardware
//
#include <hip/hip_runtime.h>
#include <math.h>

// ---------------------------------------------------------------------------
// MemoryGate for MI455X (gfx1250): bf16 WMMA GEMM pipeline + product-key topk
// GEMMs use v_wmma_f32_16x16x32_bf16 with async global->LDS tile staging
// (GLOBAL_LOAD_ASYNC_TO_LDS_B128 + s_wait_asynccnt, double-buffered).
// Wave tile 64x64 -> 1:1 ds_load_b128 : wmma ratio.
// ---------------------------------------------------------------------------

typedef __attribute__((ext_vector_type(16))) __bf16 v16bf;
typedef __attribute__((ext_vector_type(8)))  float  v8f;

#define TOK   4096      // 2 * 2048 tokens
#define DIM   1024
#define HID   2048
#define KD    512
#define HD    256
#define NKEY  1024
#define NCAND 32

#define LDSTRIDE 72     // 64 + 8 pad, ushorts (144 B, multiple of 16)

struct U8 { unsigned int u[8]; };

__device__ __forceinline__ unsigned short f2bf(float f) {
  unsigned int u = __builtin_bit_cast(unsigned int, f);
  u += 0x7FFFu + ((u >> 16) & 1u);      // round to nearest even
  return (unsigned short)(u >> 16);
}

// Load one 16x32 bf16 fragment (A or B operand) from an LDS tile whose rows
// (M for A, N for B) have LDSTRIDE ushorts and K-contiguous layout.
// ISA 16-bit A layout: lanes 0-15 -> K base 0, lanes 16-31 -> K base 8;
// VGPR v in 0..3 holds K = kb+2v..+1, v in 4..7 holds K = kb+16+2(v-4)..+1.
__device__ __forceinline__ v16bf load_frag(const unsigned short* base, int row) {
  const int lane = threadIdx.x & 31;
  const unsigned short* p = base + (size_t)(row + (lane & 15)) * LDSTRIDE + ((lane >> 4) << 3);
  U8 f;
  f.u[0] = *(const unsigned int*)(p + 0);
  f.u[1] = *(const unsigned int*)(p + 2);
  f.u[2] = *(const unsigned int*)(p + 4);
  f.u[3] = *(const unsigned int*)(p + 6);
  f.u[4] = *(const unsigned int*)(p + 16);
  f.u[5] = *(const unsigned int*)(p + 18);
  f.u[6] = *(const unsigned int*)(p + 20);
  f.u[7] = *(const unsigned int*)(p + 22);
  return __builtin_bit_cast(v16bf, f);
}

// Issue one 16B-per-lane async copy global -> LDS (ASYNCcnt tracked).
__device__ __forceinline__ void async_b128(unsigned lds_addr, const void* gaddr) {
  asm volatile("global_load_async_to_lds_b128 %0, %1, off"
               :: "v"(lds_addr), "v"(gaddr) : "memory");
}

// ---------------------------------------------------------------------------
// Generic bf16 GEMM: C[M,N] = A[M,K] (bf16, row-major) x Bt[N,K] (bf16, row-
// major, i.e. B transposed).  128x256 block, BK=64, double-buffered async LDS
// staging. 8 waves (2x4), each wave 64x64 via 4x4x2 v_wmma_f32_16x16x32_bf16.
// mode 0: C = acc                    -> f32 Cf
// mode 1: C = gelu(acc + bias)       -> bf16 Cb
// mode 2: C = acc + bias + resid     -> f32 Cf
// mode 3: C = acc * min(exp(*s),100) -> f32 Cf
// Requires M%128==0, N%256==0, K%64==0 (true for all calls here).
// ---------------------------------------------------------------------------
__global__ __launch_bounds__(256)
void gemm_bf16_kernel(const unsigned short* __restrict__ A,
                      const unsigned short* __restrict__ Bt,
                      const float* __restrict__ bias,
                      const float* __restrict__ resid,
                      const float* __restrict__ scale_ptr,
                      float* __restrict__ Cf,
                      unsigned short* __restrict__ Cb,
                      int M, int N, int K, int mode)
{
  __shared__ unsigned short As[2][128 * LDSTRIDE];
  __shared__ unsigned short Bs[2][256 * LDSTRIDE];

  const int tid  = threadIdx.x;
  const int wave = tid >> 5;
  const int wm   = (wave & 1) * 64;      // wave M offset in block
  const int wn   = (wave >> 1) * 64;     // wave N offset in block
  const int bM   = blockIdx.y * 128;
  const int bN   = blockIdx.x * 256;
  const int lrow = tid >> 1;             // 0..127, A-tile row staged by this thread
  const int lsegE = (tid & 1) * 32;      // ushort offset within A row (0 / 32)
  const int lsegB = (tid & 1) * 64;      // byte offset within LDS A row

  // Raw LDS byte offsets (generic shared pointer keeps offset in addr[31:0]).
  const unsigned aBase0 = (unsigned)(unsigned long long)(const void*)&As[0][0];
  const unsigned aBase1 = (unsigned)(unsigned long long)(const void*)&As[1][0];
  const unsigned bBase0 = (unsigned)(unsigned long long)(const void*)&Bs[0][0];
  const unsigned bBase1 = (unsigned)(unsigned long long)(const void*)&Bs[1][0];

  const unsigned aRowOff = (unsigned)lrow * (LDSTRIDE * 2) + (unsigned)lsegB;
  const unsigned bRowOff = (unsigned)tid * (LDSTRIDE * 2);  // B: one full row per thread
  const unsigned short* gA = A  + (size_t)(bM + lrow) * K + lsegE;
  const unsigned short* gB = Bt + (size_t)(bN + tid) * K;   // row `tid` of B tile

  const int nt = K >> 6;

  // prologue: stage tile 0 into buffer 0 (A: 4 chunks, B: 8 chunks per thread)
  {
    const unsigned la = aBase0 + aRowOff;
    const unsigned lb = bBase0 + bRowOff;
#pragma unroll
    for (int c = 0; c < 4; ++c) async_b128(la + c * 16, gA + c * 8);
#pragma unroll
    for (int c = 0; c < 8; ++c) async_b128(lb + c * 16, gB + c * 8);
  }

  v8f acc[4][4] = {};

  for (int t = 0; t < nt; ++t) {
    const int b = t & 1;
    if (t + 1 < nt) {
      // prefetch tile t+1 into the other buffer, then wait only for tile t
      const unsigned la = (b ? aBase0 : aBase1) + aRowOff;
      const unsigned lb = (b ? bBase0 : bBase1) + bRowOff;
      const unsigned short* pa = gA + (size_t)(t + 1) * 64;
      const unsigned short* pb = gB + (size_t)(t + 1) * 64;
#pragma unroll
      for (int c = 0; c < 4; ++c) async_b128(la + c * 16, pa + c * 8);
#pragma unroll
      for (int c = 0; c < 8; ++c) async_b128(lb + c * 16, pb + c * 8);
      asm volatile("s_wait_asynccnt 12" ::: "memory");
    } else {
      asm volatile("s_wait_asynccnt 0" ::: "memory");
    }
    __syncthreads();   // tile t resident in LDS for all waves

#pragma unroll
    for (int kh = 0; kh < 2; ++kh) {
      const unsigned short* at = &As[b][kh * 32];
      const unsigned short* bt = &Bs[b][kh * 32];
      v16bf bfr[4];
#pragma unroll
      for (int j = 0; j < 4; ++j) bfr[j] = load_frag(bt, wn + j * 16);
#pragma unroll
      for (int i = 0; i < 4; ++i) {
        v16bf af = load_frag(at, wm + i * 16);
#pragma unroll
        for (int j = 0; j < 4; ++j)
          acc[i][j] = __builtin_amdgcn_wmma_f32_16x16x32_bf16(
              false, af, false, bfr[j], (short)0, acc[i][j], false, false);
      }
    }
    __syncthreads();   // all waves done reading buffer b before it is rewritten
  }

  float scl = 1.0f;
  if (mode == 3) scl = fminf(expf(scale_ptr[0]), 100.0f);

  const int lane = tid & 31;
  const int n_l  = lane & 15;
  const int m_hi = (lane >> 4) * 8;

#pragma unroll
  for (int i = 0; i < 4; ++i) {
#pragma unroll
    for (int j = 0; j < 4; ++j) {
      const int row0 = bM + wm + i * 16 + m_hi;
      const int col  = bN + wn + j * 16 + n_l;
#pragma unroll
      for (int d = 0; d < 8; ++d) {
        const int r = row0 + d;
        float v = acc[i][j][d];
        size_t o = (size_t)r * N + col;
        if (mode == 0) {
          Cf[o] = v;
        } else if (mode == 1) {
          v += bias[col];
          v = 0.5f * v * (1.0f + erff(v * 0.70710678118654752f));  // exact GELU
          Cb[o] = f2bf(v);
        } else if (mode == 2) {
          Cf[o] = v + bias[col] + resid[o];
        } else {
          Cf[o] = v * scl;
        }
      }
    }
  }
}

// ---------------------------------------------------------------------------
// Weight transpose + f32 -> bf16:  Wt[n*K + k] = bf16(W[k*N + n])
// ---------------------------------------------------------------------------
__global__ void wtrans_kernel(const float* __restrict__ W, unsigned short* __restrict__ Wt,
                              int K, int N) {
  int idx = blockIdx.x * blockDim.x + threadIdx.x;
  if (idx >= K * N) return;
  int n = idx / K, k = idx - n * K;
  Wt[idx] = f2bf(W[(size_t)k * N + n]);
}

__global__ void cvt_kernel(const float* __restrict__ in, unsigned short* __restrict__ out, int n) {
  int idx = blockIdx.x * blockDim.x + threadIdx.x;
  if (idx < n) out[idx] = f2bf(in[idx]);
}

// ---------------------------------------------------------------------------
// LayerNorm over D=1024, output bf16.  One block (256 thr) per row.
// ---------------------------------------------------------------------------
__global__ __launch_bounds__(256)
void ln_kernel(const float* __restrict__ in, const float* __restrict__ g,
               const float* __restrict__ b, unsigned short* __restrict__ out) {
  __shared__ float red[256];
  __shared__ float mu_s, rs_s;
  const int tid = threadIdx.x;
  const float* r = in + (size_t)blockIdx.x * DIM;
  float s = 0.f, s2 = 0.f;
  for (int i = tid; i < DIM; i += 256) { float v = r[i]; s += v; s2 += v * v; }
  red[tid] = s; __syncthreads();
  for (int st = 128; st > 0; st >>= 1) { if (tid < st) red[tid] += red[tid + st]; __syncthreads(); }
  if (tid == 0) mu_s = red[0] * (1.0f / DIM);
  __syncthreads();
  red[tid] = s2; __syncthreads();
  for (int st = 128; st > 0; st >>= 1) { if (tid < st) red[tid] += red[tid + st]; __syncthreads(); }
  if (tid == 0) {
    float mu = mu_s;
    float var = red[0] * (1.0f / DIM) - mu * mu;
    rs_s = rsqrtf(var + 1e-5f);
  }
  __syncthreads();
  const float mu = mu_s, rs = rs_s;
  unsigned short* o = out + (size_t)blockIdx.x * DIM;
  for (int i = tid; i < DIM; i += 256) o[i] = f2bf((r[i] - mu) * rs * g[i] + b[i]);
}

// ---------------------------------------------------------------------------
// L2-normalize the two 256-d halves of q[4096][512] -> bf16 q1n, q2n.
// ---------------------------------------------------------------------------
__global__ __launch_bounds__(256)
void l2q_kernel(const float* __restrict__ q, unsigned short* __restrict__ q1n,
                unsigned short* __restrict__ q2n) {
  __shared__ float red[256];
  __shared__ float n0_s, n1_s;
  const int tid = threadIdx.x;
  const float* r = q + (size_t)blockIdx.x * KD;
  float v0 = r[tid], v1 = r[HD + tid];
  red[tid] = v0 * v0; __syncthreads();
  for (int st = 128; st > 0; st >>= 1) { if (tid < st) red[tid] += red[tid + st]; __syncthreads(); }
  if (tid == 0) n0_s = fmaxf(sqrtf(red[0]), 1e-12f);
  __syncthreads();
  red[tid] = v1 * v1; __syncthreads();
  for (int st = 128; st > 0; st >>= 1) { if (tid < st) red[tid] += red[tid + st]; __syncthreads(); }
  if (tid == 0) n1_s = fmaxf(sqrtf(red[0]), 1e-12f);
  __syncthreads();
  q1n[(size_t)blockIdx.x * HD + tid] = f2bf(v0 / n0_s);
  q2n[(size_t)blockIdx.x * HD + tid] = f2bf(v1 / n1_s);
}

// ---------------------------------------------------------------------------
// L2-normalize keys[2][1024][256] -> bf16 k1n[1024][256], k2n[1024][256]
// (row-major normalized keys == Bt layout for the score GEMM).
// ---------------------------------------------------------------------------
__global__ __launch_bounds__(256)
void lkeys_kernel(const float* __restrict__ keys, unsigned short* __restrict__ k1n,
                  unsigned short* __restrict__ k2n) {
  __shared__ float red[256];
  __shared__ float n_s;
  const int tid = threadIdx.x;
  const int set = blockIdx.x >> 10;
  const int key = blockIdx.x & (NKEY - 1);
  const float* p = keys + ((size_t)set * NKEY + key) * HD;
  float v = p[tid];
  red[tid] = v * v; __syncthreads();
  for (int st = 128; st > 0; st >>= 1) { if (tid < st) red[tid] += red[tid + st]; __syncthreads(); }
  if (tid == 0) n_s = fmaxf(sqrtf(red[0]), 1e-12f);
  __syncthreads();
  unsigned short* o = (set ? k2n : k1n) + (size_t)key * HD;
  o[tid] = f2bf(v / n_s);
}

// ---------------------------------------------------------------------------
// Dual top-32 + cartesian top-32 + softmax.
// Top-32 of s1[i]+s2[j] over the 128x128 rank grid equals top-32 over the
// 32x32 rank grid (k-th largest sum uses ranks <= k), so top-32 per set
// suffices and matches the reference (lexicographic tie order preserved).
// ---------------------------------------------------------------------------
__device__ __forceinline__ void block_argmax(float* vals, int L, float* rv, int* ri,
                                             int tid, float& outv, int& outi) {
  float bv = -INFINITY; int bi = 0x7FFFFFFF;
  for (int i = tid; i < L; i += 256) {
    float v = vals[i];
    if (v > bv || (v == bv && i < bi)) { bv = v; bi = i; }
  }
  rv[tid] = bv; ri[tid] = bi;
  __syncthreads();
  for (int s = 128; s > 0; s >>= 1) {
    if (tid < s) {
      float v = rv[tid + s]; int ix = ri[tid + s];
      if (v > rv[tid] || (v == rv[tid] && ix < ri[tid])) { rv[tid] = v; ri[tid] = ix; }
    }
    __syncthreads();
  }
  outv = rv[0]; outi = ri[0];
  __syncthreads();
}

__global__ __launch_bounds__(256)
void topk_kernel(const float* __restrict__ s1, const float* __restrict__ s2,
                 int* __restrict__ out_idx, float* __restrict__ out_sc) {
  __shared__ float v1[NKEY];
  __shared__ float v2[NKEY];
  __shared__ float rv[256];
  __shared__ int   ri[256];
  __shared__ float t1v[32], t2v[32], cv[32], ex[32];
  __shared__ int   t1i[32], t2i[32], ci[32];
  __shared__ float sum_s;

  const int tid = threadIdx.x;
  const int row = blockIdx.x;
  const float* p1 = s1 + (size_t)row * NKEY;
  const float* p2 = s2 + (size_t)row * NKEY;
  for (int i = tid; i < NKEY; i += 256) { v1[i] = p1[i]; v2[i] = p2[i]; }
  __syncthreads();

  for (int k = 0; k < 32; ++k) {
    float bv; int bi;
    block_argmax(v1, NKEY, rv, ri, tid, bv, bi);
    if (tid == 0) { t1v[k] = bv; t1i[k] = bi; v1[bi] = -INFINITY; }
    __syncthreads();
  }
  for (int k = 0; k < 32; ++k) {
    float bv; int bi;
    block_argmax(v2, NKEY, rv, ri, tid, bv, bi);
    if (tid == 0) { t2v[k] = bv; t2i[k] = bi; v2[bi] = -INFINITY; }
    __syncthreads();
  }
  for (int i = tid; i < 1024; i += 256) v1[i] = t1v[i >> 5] + t2v[i & 31];
  __syncthreads();
  for (int k = 0; k < 32; ++k) {
    float bv; int bi;
    block_argmax(v1, 1024, rv, ri, tid, bv, bi);
    if (tid == 0) { cv[k] = bv; ci[k] = bi; v1[bi] = -INFINITY; }
    __syncthreads();
  }
  if (tid < 32) ex[tid] = expf(cv[tid] - cv[0]);
  __syncthreads();
  if (tid == 0) {
    float s = 0.f;
    for (int k = 0; k < 32; ++k) s += ex[k];
    sum_s = s;
  }
  __syncthreads();
  if (tid < 32) {
    int pi = ci[tid];
    out_idx[(size_t)row * NCAND + tid] = t1i[pi >> 5] * NKEY + t2i[pi & 31];
    out_sc [(size_t)row * NCAND + tid] = ex[tid] / sum_s;
  }
}

// ---------------------------------------------------------------------------
// Host-side orchestration
// ---------------------------------------------------------------------------
static inline size_t align_up(size_t x) { return (x + 255) & ~(size_t)255; }

extern "C" void kernel_launch(void* const* d_in, const int* in_sizes, int n_in,
                              void* d_out, int out_size, void* d_ws, size_t ws_size,
                              hipStream_t stream) {
  const float* x     = (const float*)d_in[0];
  const float* w_in  = (const float*)d_in[1];
  const float* ln1_g = (const float*)d_in[2];
  const float* ln1_b = (const float*)d_in[3];
  const float* w1a   = (const float*)d_in[4];
  const float* b1a   = (const float*)d_in[5];
  const float* w1b   = (const float*)d_in[6];
  const float* b1b   = (const float*)d_in[7];
  const float* ln2_g = (const float*)d_in[8];
  const float* ln2_b = (const float*)d_in[9];
  const float* w2a   = (const float*)d_in[10];
  const float* b2a   = (const float*)d_in[11];
  const float* w2b   = (const float*)d_in[12];
  const float* b2b   = (const float*)d_in[13];
  const float* lnf_g = (const float*)d_in[14];
  const float* lnf_b = (const float*)d_in[15];
  const float* w_out = (const float*)d_in[16];
  const float* keys  = (const float*)d_in[17];
  const float* lsc   = (const float*)d_in[18];

  char* ws = (char*)d_ws;
  size_t off = 0;
  auto take = [&](size_t bytes) { char* p = ws + off; off = align_up(off + bytes); return p; };

  unsigned short* w_inT = (unsigned short*)take((size_t)DIM * DIM * 2);
  unsigned short* w1aT  = (unsigned short*)take((size_t)DIM * HID * 2);
  unsigned short* w1bT  = (unsigned short*)take((size_t)HID * DIM * 2);
  unsigned short* w2aT  = (unsigned short*)take((size_t)DIM * HID * 2);
  unsigned short* w2bT  = (unsigned short*)take((size_t)HID * DIM * 2);
  unsigned short* w_outT= (unsigned short*)take((size_t)DIM * KD * 2);
  unsigned short* k1n   = (unsigned short*)take((size_t)NKEY * HD * 2);
  unsigned short* k2n   = (unsigned short*)take((size_t)NKEY * HD * 2);
  unsigned short* actA  = (unsigned short*)take((size_t)TOK * DIM * 2);   // x / LN outputs
  unsigned short* actB  = (unsigned short*)take((size_t)TOK * HID * 2);   // hidden GELU acts
  float* h0 = (float*)take((size_t)TOK * DIM * 4);   // also reused for h2
  float* h1 = (float*)take((size_t)TOK * DIM * 4);
  float* qf = (float*)take((size_t)TOK * KD * 4);
  unsigned short* q1n = (unsigned short*)take((size_t)TOK * HD * 2);
  unsigned short* q2n = (unsigned short*)take((size_t)TOK * HD * 2);
  float* sc1 = (float*)take((size_t)TOK * NKEY * 4);
  float* sc2 = (float*)take((size_t)TOK * NKEY * 4);
  (void)ws_size; (void)in_sizes; (void)n_in; (void)out_size;

  const int T = 256;
  auto blocks = [&](int n) { return (n + T - 1) / T; };

  // weight preparation (transpose + bf16)
  wtrans_kernel<<<blocks(DIM * DIM), T, 0, stream>>>(w_in, w_inT, DIM, DIM);
  wtrans_kernel<<<blocks(DIM * HID), T, 0, stream>>>(w1a, w1aT, DIM, HID);
  wtrans_kernel<<<blocks(HID * DIM), T, 0, stream>>>(w1b, w1bT, HID, DIM);
  wtrans_kernel<<<blocks(DIM * HID), T, 0, stream>>>(w2a, w2aT, DIM, HID);
  wtrans_kernel<<<blocks(HID * DIM), T, 0, stream>>>(w2b, w2bT, HID, DIM);
  wtrans_kernel<<<blocks(DIM * KD),  T, 0, stream>>>(w_out, w_outT, DIM, KD);
  cvt_kernel<<<blocks(TOK * DIM), T, 0, stream>>>(x, actA, TOK * DIM);
  lkeys_kernel<<<2 * NKEY, T, 0, stream>>>(keys, k1n, k2n);

  // h0 = x @ w_in
  gemm_bf16_kernel<<<dim3(DIM / 256, TOK / 128), T, 0, stream>>>(
      actA, w_inT, nullptr, nullptr, nullptr, h0, nullptr, TOK, DIM, DIM, 0);
  // block 1
  ln_kernel<<<TOK, T, 0, stream>>>(h0, ln1_g, ln1_b, actA);
  gemm_bf16_kernel<<<dim3(HID / 256, TOK / 128), T, 0, stream>>>(
      actA, w1aT, b1a, nullptr, nullptr, nullptr, actB, TOK, HID, DIM, 1);
  gemm_bf16_kernel<<<dim3(DIM / 256, TOK / 128), T, 0, stream>>>(
      actB, w1bT, b1b, h0, nullptr, h1, nullptr, TOK, DIM, HID, 2);
  // block 2 (h2 reuses h0's buffer)
  ln_kernel<<<TOK, T, 0, stream>>>(h1, ln2_g, ln2_b, actA);
  gemm_bf16_kernel<<<dim3(HID / 256, TOK / 128), T, 0, stream>>>(
      actA, w2aT, b2a, nullptr, nullptr, nullptr, actB, TOK, HID, DIM, 1);
  gemm_bf16_kernel<<<dim3(DIM / 256, TOK / 128), T, 0, stream>>>(
      actB, w2bT, b2b, h1, nullptr, h0, nullptr, TOK, DIM, HID, 2);
  // final LN + projection to KDIM
  ln_kernel<<<TOK, T, 0, stream>>>(h0, lnf_g, lnf_b, actA);
  gemm_bf16_kernel<<<dim3(KD / 256, TOK / 128), T, 0, stream>>>(
      actA, w_outT, nullptr, nullptr, nullptr, qf, nullptr, TOK, KD, DIM, 0);
  // normalize query halves, score against normalized keys (scaled)
  l2q_kernel<<<TOK, T, 0, stream>>>(qf, q1n, q2n);
  gemm_bf16_kernel<<<dim3(NKEY / 256, TOK / 128), T, 0, stream>>>(
      q1n, k1n, nullptr, nullptr, lsc, sc1, nullptr, TOK, NKEY, HD, 3);
  gemm_bf16_kernel<<<dim3(NKEY / 256, TOK / 128), T, 0, stream>>>(
      q2n, k2n, nullptr, nullptr, lsc, sc2, nullptr, TOK, NKEY, HD, 3);

  // dual top-k + combine + softmax; outputs: [idx int32 x 131072][scores f32 x 131072]
  int*   out_idx = (int*)d_out;
  float* out_sc  = (float*)d_out + (size_t)TOK * NCAND;
  topk_kernel<<<TOK, T, 0, stream>>>(sc1, sc2, out_idx, out_sc);
}
